// MultiHeadSpatioTemporalAttention_4621384810729
// MI455X (gfx1250) — matrix operand order repaired and checked by reading, not compiled, not run on hardware
//
#include <hip/hip_runtime.h>
#include <hip/hip_bf16.h>
#include <cstdint>
#include <cstddef>

// ---------------------------------------------------------------------------
// MHA block for MI455X (gfx1250, wave32, WMMA 16x16x32 f16->f32)
// y, attn = LN(x + (softmax(QK^T/sqrt(dk)) V) Wo^T + bo), shapes:
//   x: [2,2048,1024], W*: [1024,1024], out y: [2,2048,1024] f32,
//   attn: [2,16,2048,2048] f32  (d_out = y ++ attn)
// Data movement: GLOBAL_LOAD_ASYNC_TO_LDS_B128 (ASYNCcnt), double-buffered LDS
// so the next K-tile's DMA overlaps the current tile's WMMA compute.
// ---------------------------------------------------------------------------

typedef _Float16 v16h __attribute__((ext_vector_type(16)));
typedef _Float16 v8h  __attribute__((ext_vector_type(8)));
typedef float    v8f  __attribute__((ext_vector_type(8)));

#define D_MODEL 1024
#define NHEADS  16
#define DK      64
#define SEQ     2048
#define BATCH   2
#define MTOT    (BATCH * SEQ)          // 4096 tokens

// ---------------------------------------------------------------------------
// Async memory -> LDS DMA, 16 bytes per lane (CDNA5 ISA 10 / 08_async_tensor).
// VDST VGPR carries the LDS byte address (flat-pointer low 32 bits == LDS
// offset per ISA 10.2 aperture mapping); VADDR carries the 64-bit global addr.
// Tracked by ASYNCcnt; async loads complete in order, so waiting
// ASYNCcnt <= N leaves the most recent N transfers in flight.
// ---------------------------------------------------------------------------
__device__ __forceinline__ void async_copy_b128(void* lds_dst, const void* gsrc) {
  const uint32_t lds_off = (uint32_t)(uintptr_t)lds_dst;
  const unsigned long long ga = (unsigned long long)(uintptr_t)gsrc;
  asm volatile("global_load_async_to_lds_b128 %0, %1, off"
               :
               : "v"(lds_off), "v"(ga)
               : "memory");
}

template <int N>
__device__ __forceinline__ void wait_async_le() {
  asm volatile("s_wait_asynccnt %0" ::"n"(N) : "memory");
}

// ---------------------------------------------------------------------------
// WMMA fragment loaders (CDNA5 ISA 7.12.2 layouts, wave32)
// A 16x32 f16: lane L -> row m = L%16; K = {h*8..h*8+7, 16+h*8..16+h*8+7}, h=L/16
// B 32x16 f16: lane L -> col n = L%16; K = {h*8..h*8+7, 16+h*8..16+h*8+7}, h=L/16
// C 16x16 f32: VGPR r -> M = r + 8*(L/16); N = L%16
// ---------------------------------------------------------------------------
__device__ __forceinline__ v16h load_a_frag(const _Float16* sA, int lda, int m0, int lane) {
  const int m = m0 + (lane & 15);
  const int h = lane >> 4;
  v8h lo = *reinterpret_cast<const v8h*>(sA + m * lda + h * 8);
  v8h hi = *reinterpret_cast<const v8h*>(sA + m * lda + 16 + h * 8);
  v16h r;
#pragma unroll
  for (int i = 0; i < 8; ++i) { r[i] = lo[i]; r[i + 8] = hi[i]; }
  return r;
}

__device__ __forceinline__ v16h load_b_frag(const _Float16* sB, int ldb, int n0, int lane) {
  const int n = n0 + (lane & 15);
  const int h = lane >> 4;
  v16h r;
#pragma unroll
  for (int i = 0; i < 8; ++i) {
    r[i] = sB[(h * 8 + i) * ldb + n];
    r[i + 8] = sB[(16 + h * 8 + i) * ldb + n];
  }
  return r;
}

// ---------------------------------------------------------------------------
// Generic tiled WMMA GEMM: C[M,N] = A[M,K] (row) x B[K,N] (row), f32 accum.
// Block tile 128 x BN x 32, 256 threads = 8 waves (2x4), wave tile 64 x (BN/4).
// Double-buffered LDS; async DMA for the next tile overlaps WMMA on current.
// MODE: 0 = f16 out, scatter [B,H,S,DK]      (Q, V projections)
//       1 = f16 out, scatter [B,H,DK,S]      (K^T projection)
//       2 = f32 logits, batched [bz][m][n]   (scores)
//       3 = f16 ctx, scatter [token][h*DK+n] (attn @ V; A is f32 probs)
//       4 = f32 out + bias + residual        (output projection)
// ---------------------------------------------------------------------------
template <int BN, int MODE, bool AF32>
__global__ __launch_bounds__(256) void gemm_wmma_k(
    const void* __restrict__ Av, const _Float16* __restrict__ Bmat,
    void* __restrict__ Cout, const float* __restrict__ bias,
    const float* __restrict__ resid,
    int lda, int ldb, int K, long batchA, long batchB) {
  constexpr int BM = 128, BK = 32;
  constexpr int NI = BN / 64;                     // n-tiles per wave (1 or 2)
  // async ops issued per thread per tile (A chunks unless f32-converted, + B)
  constexpr int NOPS = (AF32 ? 0 : 2) + BN / 64;
  __shared__ alignas(32) _Float16 sA[2][BM * BK];
  __shared__ alignas(32) _Float16 sB[2][BK * BN];

  const int tid  = threadIdx.x;
  const int lane = tid & 31;
  const int wid  = tid >> 5;
  const int bm   = blockIdx.y * BM;
  const int bn   = blockIdx.x * BN;
  const int bz   = blockIdx.z;
  const int wm   = (wid & 1) * 64;
  const int wn   = (wid >> 1) * (NI * 16);

  const _Float16* Ah = nullptr;
  const float*    Af = nullptr;
  if constexpr (AF32) Af = reinterpret_cast<const float*>(Av) + (size_t)bz * batchA;
  else                Ah = reinterpret_cast<const _Float16*>(Av) + (size_t)bz * batchA;
  const _Float16* Bp = Bmat + (size_t)bz * batchB;

  // ---- stage one K-tile into LDS buffer `buf` ----
  auto stage_tile = [&](int kt, int buf) {
#pragma unroll
    for (int t = 0; t < 2; ++t) {                 // A: 128x32 f16, 512 chunks
      const int c   = tid + t * 256;
      const int row = c >> 2;
      const int col = (c & 3) * 8;
      if constexpr (AF32) {
        const float* src = Af + (size_t)(bm + row) * lda + kt * BK + col;
#pragma unroll
        for (int i = 0; i < 8; ++i) sA[buf][row * BK + col + i] = (_Float16)src[i];
      } else {
        async_copy_b128(&sA[buf][row * BK + col],
                        Ah + (size_t)(bm + row) * lda + kt * BK + col);
      }
    }
    constexpr int CPR = BN / 8;                   // B: 32xBN f16
#pragma unroll
    for (int t = 0; t < BN / 64; ++t) {
      const int c   = tid + t * 256;
      const int row = c / CPR;
      const int col = (c % CPR) * 8;
      async_copy_b128(&sB[buf][row * BN + col],
                      Bp + (size_t)(kt * BK + row) * ldb + bn + col);
    }
  };

  v8f acc[4][NI];
#pragma unroll
  for (int mi = 0; mi < 4; ++mi)
#pragma unroll
    for (int ni = 0; ni < NI; ++ni)
#pragma unroll
      for (int e = 0; e < 8; ++e) acc[mi][ni][e] = 0.0f;

  const int nk = K / BK;
  stage_tile(0, 0);                               // prologue DMA
  for (int kt = 0; kt < nk; ++kt) {
    const int cur = kt & 1;
    __syncthreads();          // everyone done reading buf[cur^1] (iter kt-1)
    if (kt + 1 < nk) {
      stage_tile(kt + 1, cur ^ 1);                // DMA next tile (in flight)
      wait_async_le<NOPS>();                      // current tile landed
    } else {
      wait_async_le<0>();
    }
    __syncthreads();          // current tile visible to all waves

    v16h afrag[4];
#pragma unroll
    for (int mi = 0; mi < 4; ++mi)
      afrag[mi] = load_a_frag(sA[cur], BK, wm + mi * 16, lane);
    v16h bfrag[NI];
#pragma unroll
    for (int ni = 0; ni < NI; ++ni)
      bfrag[ni] = load_b_frag(sB[cur], BN, wn + ni * 16, lane);
#pragma unroll
    for (int mi = 0; mi < 4; ++mi)
#pragma unroll
      for (int ni = 0; ni < NI; ++ni)
        acc[mi][ni] = __builtin_amdgcn_wmma_f32_16x16x32_f16(
            false, afrag[mi], false, bfrag[ni], (short)0, acc[mi][ni], false, false);
  }

  // ---- epilogue ----
  float*    Cf = reinterpret_cast<float*>(Cout);
  _Float16* Ch = reinterpret_cast<_Float16*>(Cout);
#pragma unroll
  for (int mi = 0; mi < 4; ++mi) {
#pragma unroll
    for (int ni = 0; ni < NI; ++ni) {
#pragma unroll
      for (int r = 0; r < 8; ++r) {
        const int m = bm + wm + mi * 16 + r + ((lane >> 4) << 3);
        const int n = bn + wn + ni * 16 + (lane & 15);
        const float v = acc[mi][ni][r];
        if constexpr (MODE == 0) {            // Q/V: [B,H,S,DK]
          const int b = m >> 11, s = m & (SEQ - 1), h = n >> 6, d = n & (DK - 1);
          Ch[((size_t)(b * NHEADS + h) * SEQ + s) * DK + d] = (_Float16)v;
        } else if constexpr (MODE == 1) {     // K^T: [B,H,DK,S]
          const int b = m >> 11, s = m & (SEQ - 1), h = n >> 6, d = n & (DK - 1);
          Ch[((size_t)(b * NHEADS + h) * DK + d) * SEQ + s] = (_Float16)v;
        } else if constexpr (MODE == 2) {     // logits f32, batched
          Cf[(size_t)bz * ((size_t)SEQ * SEQ) + (size_t)m * SEQ + n] = v;
        } else if constexpr (MODE == 3) {     // ctx f16: [token][h*DK+n]
          const int b = bz >> 4, h = bz & 15;
          Ch[((size_t)(b * SEQ + m)) * D_MODEL + h * DK + n] = (_Float16)v;
        } else {                              // out proj + bias + residual
          const size_t idx = (size_t)m * D_MODEL + n;
          Cf[idx] = v + bias[n] + resid[idx];
        }
      }
    }
  }
}

// ---------------------------------------------------------------------------
// Row softmax over 2048 logits, in place; applies 1/sqrt(DK) scale.
// ---------------------------------------------------------------------------
__global__ __launch_bounds__(256) void softmax_rows_k(float* __restrict__ attn) {
  __shared__ float red[256];
  const int t = threadIdx.x;
  float* p = attn + (size_t)blockIdx.x * SEQ;

  float vals[8];
  float mx = -3.4e38f;
#pragma unroll
  for (int i = 0; i < 8; ++i) {
    vals[i] = p[t + i * 256] * 0.125f;      // 1/sqrt(64)
    mx = fmaxf(mx, vals[i]);
  }
  red[t] = mx;
  __syncthreads();
  for (int s = 128; s > 0; s >>= 1) {
    if (t < s) red[t] = fmaxf(red[t], red[t + s]);
    __syncthreads();
  }
  mx = red[0];
  __syncthreads();

  float sm = 0.0f;
#pragma unroll
  for (int i = 0; i < 8; ++i) {
    vals[i] = __expf(vals[i] - mx);
    sm += vals[i];
  }
  red[t] = sm;
  __syncthreads();
  for (int s = 128; s > 0; s >>= 1) {
    if (t < s) red[t] += red[t + s];
    __syncthreads();
  }
  const float inv = 1.0f / red[0];
#pragma unroll
  for (int i = 0; i < 8; ++i) p[t + i * 256] = vals[i] * inv;
}

// ---------------------------------------------------------------------------
// Per-token LayerNorm over 1024 features.
// ---------------------------------------------------------------------------
__global__ __launch_bounds__(256) void layernorm_rows_k(
    const float* __restrict__ y, const float* __restrict__ gamma,
    const float* __restrict__ beta, float* __restrict__ out) {
  __shared__ float r1[256];
  __shared__ float r2[256];
  const int t = threadIdx.x;
  const float* p = y + (size_t)blockIdx.x * D_MODEL;

  float v[4], s = 0.0f, ss = 0.0f;
#pragma unroll
  for (int i = 0; i < 4; ++i) {
    v[i] = p[t + i * 256];
    s += v[i];
    ss += v[i] * v[i];
  }
  r1[t] = s; r2[t] = ss;
  __syncthreads();
  for (int k = 128; k > 0; k >>= 1) {
    if (t < k) { r1[t] += r1[t + k]; r2[t] += r2[t + k]; }
    __syncthreads();
  }
  const float mu  = r1[0] * (1.0f / D_MODEL);
  const float var = r2[0] * (1.0f / D_MODEL) - mu * mu;
  const float rs  = rsqrtf(var + 1e-5f);
  float* o = out + (size_t)blockIdx.x * D_MODEL;
#pragma unroll
  for (int i = 0; i < 4; ++i) {
    const int c = t + i * 256;
    o[c] = (v[i] - mu) * rs * gamma[c] + beta[c];
  }
}

// ---------------------------------------------------------------------------
// Conversions
// ---------------------------------------------------------------------------
__global__ __launch_bounds__(256) void cvt_f32_f16_k(const float* __restrict__ in,
                                                     _Float16* __restrict__ out) {
  const size_t i = (size_t)blockIdx.x * 256 + threadIdx.x;
  out[i] = (_Float16)in[i];
}

// W [N=1024][K=1024] f32 row-major -> Wt [K][N] f16 (so GEMM B is [K][N])
__global__ __launch_bounds__(256) void cvt_transpose_f16_k(const float* __restrict__ W,
                                                           _Float16* __restrict__ Wt) {
  const size_t i = (size_t)blockIdx.x * 256 + threadIdx.x;
  const int n = (int)(i >> 10);
  const int k = (int)(i & 1023);
  Wt[(size_t)k * D_MODEL + n] = (_Float16)W[i];
}

// ---------------------------------------------------------------------------
// Host-side launch
// ---------------------------------------------------------------------------
extern "C" void kernel_launch(void* const* d_in, const int* in_sizes, int n_in,
                              void* d_out, int out_size, void* d_ws, size_t ws_size,
                              hipStream_t stream) {
  const float* x     = (const float*)d_in[0];
  const float* Wq    = (const float*)d_in[1];
  const float* Wk    = (const float*)d_in[2];
  const float* Wv    = (const float*)d_in[3];
  const float* Wo    = (const float*)d_in[4];
  const float* bo    = (const float*)d_in[5];
  const float* gamma = (const float*)d_in[6];
  const float* beta  = (const float*)d_in[7];

  float* yout  = (float*)d_out;                       // [4096,1024]
  float* attnF = yout + (size_t)MTOT * D_MODEL;       // [2,16,2048,2048]

  // workspace layout (bytes)
  char* w = (char*)d_ws;
  _Float16* xh   = (_Float16*)(w + 0);                //  8 MB  [4096,1024]
  _Float16* WqT  = (_Float16*)(w + 8388608);          //  2 MB  [K][N]
  _Float16* WkT  = (_Float16*)(w + 10485760);
  _Float16* WvT  = (_Float16*)(w + 12582912);
  _Float16* WoT  = (_Float16*)(w + 14680064);
  _Float16* Qh   = (_Float16*)(w + 16777216);         //  8 MB  [B,H,S,DK]
  _Float16* Kt   = (_Float16*)(w + 25165824);         //  8 MB  [B,H,DK,S]
  _Float16* Vh   = (_Float16*)(w + 33554432);         //  8 MB  [B,H,S,DK]
  _Float16* ctxh = (_Float16*)(w + 41943040);         //  8 MB  [4096,1024]
  float*    ybuf = (float*)   (w + 50331648);         // 16 MB  [4096,1024]

  // 1) convert inputs to f16 (weights transposed to [K][N])
  cvt_f32_f16_k<<<(MTOT * D_MODEL) / 256, 256, 0, stream>>>(x, xh);
  cvt_transpose_f16_k<<<(D_MODEL * D_MODEL) / 256, 256, 0, stream>>>(Wq, WqT);
  cvt_transpose_f16_k<<<(D_MODEL * D_MODEL) / 256, 256, 0, stream>>>(Wk, WkT);
  cvt_transpose_f16_k<<<(D_MODEL * D_MODEL) / 256, 256, 0, stream>>>(Wv, WvT);
  cvt_transpose_f16_k<<<(D_MODEL * D_MODEL) / 256, 256, 0, stream>>>(Wo, WoT);

  // 2) QKV projections: [4096,1024] x [1024,1024]
  dim3 gProj(D_MODEL / 128, MTOT / 128, 1);
  gemm_wmma_k<128, 0, false><<<gProj, 256, 0, stream>>>(
      xh, WqT, Qh, nullptr, nullptr, D_MODEL, D_MODEL, D_MODEL, 0, 0);
  gemm_wmma_k<128, 1, false><<<gProj, 256, 0, stream>>>(
      xh, WkT, Kt, nullptr, nullptr, D_MODEL, D_MODEL, D_MODEL, 0, 0);
  gemm_wmma_k<128, 0, false><<<gProj, 256, 0, stream>>>(
      xh, WvT, Vh, nullptr, nullptr, D_MODEL, D_MODEL, D_MODEL, 0, 0);

  // 3) scores = Q K^T per (b,h): [2048,64] x [64,2048] -> f32 logits
  dim3 gScores(SEQ / 128, SEQ / 128, BATCH * NHEADS);
  gemm_wmma_k<128, 2, false><<<gScores, 256, 0, stream>>>(
      Qh, Kt, attnF, nullptr, nullptr, DK, SEQ, DK,
      (long)SEQ * DK, (long)DK * SEQ);

  // 4) softmax rows, in place (writes attn output)
  softmax_rows_k<<<BATCH * NHEADS * SEQ, 256, 0, stream>>>(attnF);

  // 5) ctx = attn @ V per (b,h): [2048,2048] x [2048,64] (A f32 -> f16 in LDS)
  dim3 gCtx(DK / 64, SEQ / 128, BATCH * NHEADS);
  gemm_wmma_k<64, 3, true><<<gCtx, 256, 0, stream>>>(
      attnF, Vh, ctxh, nullptr, nullptr, SEQ, DK, SEQ,
      (long)SEQ * SEQ, (long)SEQ * DK);

  // 6) out = ctx @ Wo^T + bo + x  (f32)
  gemm_wmma_k<128, 4, false><<<gProj, 256, 0, stream>>>(
      ctxh, WoT, ybuf, bo, x, D_MODEL, D_MODEL, D_MODEL, 0, 0);

  // 7) LayerNorm -> y output
  layernorm_rows_k<<<MTOT, 256, 0, stream>>>(ybuf, gamma, beta, yout);
}
